// AtomicComposition_19121194402015
// MI455X (gfx1250) — compile-verified, matrix operand order checked by reading, not run on hardware
//
#include <hip/hip_runtime.h>
#include <stdint.h>

typedef __attribute__((ext_vector_type(16))) int v16i;
typedef __attribute__((ext_vector_type(8))) float v8f;
typedef int v4i __attribute__((vector_size(16)));  // async builtin param type

#define WAVES_PER_BLOCK 8
#define BLOCK_THREADS (WAVES_PER_BLOCK * 32)

// gfx1250 async VMEM->LDS path (builtins confirmed present in rounds 1-3)
#if defined(__HIP_DEVICE_COMPILE__) && defined(__gfx1250__)
#if __has_builtin(__builtin_amdgcn_global_load_async_to_lds_b128) && \
    __has_builtin(__builtin_amdgcn_s_wait_asynccnt)
#define USE_ASYNC_LDS 1
#endif
#endif
#ifndef USE_ASYNC_LDS
#define USE_ASYNC_LDS 0
#endif

// One-hot over 4 packed bin-bytes (each byte in [0,15]):
// result byte = FP8 E4M3 2^-6 (0x08) iff bin byte == N, else 0x00.
// Paired with A bytes = FP8 64.0 (0x68): each match contributes exactly 1.0.
__device__ __forceinline__ unsigned onehot_fp8(unsigned bins4, unsigned nRep) {
  unsigned z = bins4 ^ nRep;  // nibble-valued bytes; 0 exactly where bin == N
  unsigned t = z | (z >> 1);
  t |= (t >> 2);              // bit0 of each byte = OR of bits 3:0 of that byte
  return (~(t << 3)) & 0x08080808u;  // byte -> 0x08 iff bin == N (no multiply)
}

__device__ __forceinline__ int bin_of(const unsigned char* lut, int z) {
  z = z < 0 ? 0 : (z > 127 ? 127 : z);  // matches reference clip(0,127)
  return (int)(lut[z] & 15);            // invalid (0xFF) -> column 15 (never read)
}

__global__ __launch_bounds__(BLOCK_THREADS) void
AtomicComposition_19121194402015_kernel(const int* __restrict__ species,
                                        const int* __restrict__ structure_offsets,
                                        const int* __restrict__ all_species,
                                        float* __restrict__ out,
                                        int n_structures, int n_atoms,
                                        int n_species) {
  __shared__ unsigned char lut[128];
#if USE_ASYNC_LDS
  __shared__ __align__(16) int spBuf[WAVES_PER_BLOCK][256];
#endif
  __shared__ __align__(16) unsigned char binBuf[WAVES_PER_BLOCK][256];

  const int tid = (int)threadIdx.x;
  if (tid < 128) lut[tid] = 0xFFu;
  __syncthreads();
  if (tid < n_species) {
    int z = all_species[tid];
    if (z >= 0 && z < 128) lut[z] = (unsigned char)tid;
  }
  __syncthreads();

  const int wave = tid >> 5;
  const int lane = tid & 31;
  const int s = (int)blockIdx.x * WAVES_PER_BLOCK + wave;
  if (s >= n_structures) return;  // wave-uniform; no barriers after this point

  const int aStart = structure_offsets[s];
  const int aEnd = (s + 1 < n_structures) ? structure_offsets[s + 1] : n_atoms;

  const int g16 = (lane >> 4) << 4;  // 0 or 16 : which K sub-half this lane holds
  const unsigned nRep = (unsigned)(lane & 15) * 0x01010101u;  // column id N

  v8f acc = {0.f, 0.f, 0.f, 0.f, 0.f, 0.f, 0.f, 0.f};
  v16i scaleA;
#pragma unroll
  for (int i = 0; i < 16; ++i) scaleA[i] = 0x68686868;  // A = FP8 64.0 (16x128)

  unsigned char* bins = &binBuf[wave][0];

  for (int base = aStart; base < aEnd; base += 256) {
    int cnt = aEnd - base;
    if (cnt > 256) cnt = 256;

    int4 sp0, sp1;
    if (((base & 3) == 0) && (base + 256 <= n_atoms)) {
#if USE_ASYNC_LDS
      // Stage the 1KB species block into LDS via the async data path
      __builtin_amdgcn_global_load_async_to_lds_b128(
          (__attribute__((address_space(1))) v4i*)(species + base + lane * 4),
          (__attribute__((address_space(3))) v4i*)(&spBuf[wave][lane * 4]),
          0, 0);
      __builtin_amdgcn_global_load_async_to_lds_b128(
          (__attribute__((address_space(1))) v4i*)(species + base + 128 + lane * 4),
          (__attribute__((address_space(3))) v4i*)(&spBuf[wave][128 + lane * 4]),
          0, 0);
      __builtin_amdgcn_s_wait_asynccnt(0);
      sp0 = *(const int4*)&spBuf[wave][lane * 4];
      sp1 = *(const int4*)&spBuf[wave][128 + lane * 4];
#else
      sp0 = *(const int4*)(species + base + lane * 4);
      sp1 = *(const int4*)(species + base + 128 + lane * 4);
#endif
    } else {  // array-end / unaligned tail: guarded scalar loads
      int i0 = base + lane * 4, i1 = base + 128 + lane * 4;
      sp0.x = (i0 + 0 < n_atoms) ? species[i0 + 0] : 0;
      sp0.y = (i0 + 1 < n_atoms) ? species[i0 + 1] : 0;
      sp0.z = (i0 + 2 < n_atoms) ? species[i0 + 2] : 0;
      sp0.w = (i0 + 3 < n_atoms) ? species[i0 + 3] : 0;
      sp1.x = (i1 + 0 < n_atoms) ? species[i1 + 0] : 0;
      sp1.y = (i1 + 1 < n_atoms) ? species[i1 + 1] : 0;
      sp1.z = (i1 + 2 < n_atoms) ? species[i1 + 2] : 0;
      sp1.w = (i1 + 3 < n_atoms) ? species[i1 + 3] : 0;
    }

    // bins[k] = species bin of atom (base+k); atoms past cnt -> 15 (dropped)
    const int k0 = lane * 4, k1 = 128 + lane * 4;
    unsigned w0 =
        (unsigned)((k0 + 0 < cnt) ? bin_of(lut, sp0.x) : 15) |
        ((unsigned)((k0 + 1 < cnt) ? bin_of(lut, sp0.y) : 15) << 8) |
        ((unsigned)((k0 + 2 < cnt) ? bin_of(lut, sp0.z) : 15) << 16) |
        ((unsigned)((k0 + 3 < cnt) ? bin_of(lut, sp0.w) : 15) << 24);
    unsigned w1 =
        (unsigned)((k1 + 0 < cnt) ? bin_of(lut, sp1.x) : 15) |
        ((unsigned)((k1 + 1 < cnt) ? bin_of(lut, sp1.y) : 15) << 8) |
        ((unsigned)((k1 + 2 < cnt) ? bin_of(lut, sp1.z) : 15) << 16) |
        ((unsigned)((k1 + 3 < cnt) ? bin_of(lut, sp1.w) : 15) << 24);
    *(unsigned*)&bins[k0] = w0;  // same-wave LDS ops are in-order (DScnt)
    *(unsigned*)&bins[k1] = w1;

    // 2 x (16x16x128 FP8 WMMA): D[m][n] += sum_k (bin_k == n) * (64 * 2^-6)
    // B layout (128x16, 8-bit, 16 VGPRs): lane holds N = lane%16;
    // quarter q (V4q..V4q+3) covers K = 32q + 16*(lane/16) + [0..15].
#pragma unroll
    for (int t = 0; t < 2; ++t) {
      const int tb = t * 128 + g16;
      const uint4 q0 = *(const uint4*)&bins[tb];        // K q=0
      const uint4 q1 = *(const uint4*)&bins[tb + 32];   // K q=1
      const uint4 q2 = *(const uint4*)&bins[tb + 64];   // K q=2
      const uint4 q3 = *(const uint4*)&bins[tb + 96];   // K q=3
      v16i B;
      B[0]  = (int)onehot_fp8(q0.x, nRep);
      B[1]  = (int)onehot_fp8(q0.y, nRep);
      B[2]  = (int)onehot_fp8(q0.z, nRep);
      B[3]  = (int)onehot_fp8(q0.w, nRep);
      B[4]  = (int)onehot_fp8(q1.x, nRep);
      B[5]  = (int)onehot_fp8(q1.y, nRep);
      B[6]  = (int)onehot_fp8(q1.z, nRep);
      B[7]  = (int)onehot_fp8(q1.w, nRep);
      B[8]  = (int)onehot_fp8(q2.x, nRep);
      B[9]  = (int)onehot_fp8(q2.y, nRep);
      B[10] = (int)onehot_fp8(q2.z, nRep);
      B[11] = (int)onehot_fp8(q2.w, nRep);
      B[12] = (int)onehot_fp8(q3.x, nRep);
      B[13] = (int)onehot_fp8(q3.y, nRep);
      B[14] = (int)onehot_fp8(q3.z, nRep);
      B[15] = (int)onehot_fp8(q3.w, nRep);
      // (A, B, c_mod, C, reuse_a, reuse_b)
      acc = __builtin_amdgcn_wmma_f32_16x16x128_fp8_fp8(scaleA, B, (short)0,
                                                        acc, false, false);
    }
  }

  // D row M=0 lives in acc[0] of lanes 0..15 (lane == N): float counts per bin.
  if (lane < n_species) out[(long)s * n_species + lane] = acc[0];
}

extern "C" void kernel_launch(void* const* d_in, const int* in_sizes, int n_in,
                              void* d_out, int out_size, void* d_ws,
                              size_t ws_size, hipStream_t stream) {
  (void)n_in; (void)out_size; (void)d_ws; (void)ws_size;
  // setup_inputs order: 0 positions, 1 cells, 2 species, 3 cell_shifts,
  // 4 centers, 5 pairs, 6 structure_centers, 7 structure_pairs,
  // 8 structure_offsets, 9 all_species
  const int* species = (const int*)d_in[2];
  const int* structure_offsets = (const int*)d_in[8];
  const int* all_species = (const int*)d_in[9];
  float* out = (float*)d_out;

  const int n_atoms = in_sizes[2];
  const int n_structures = in_sizes[8];
  const int n_species = in_sizes[9];

  const int blocks = (n_structures + WAVES_PER_BLOCK - 1) / WAVES_PER_BLOCK;
  AtomicComposition_19121194402015_kernel<<<dim3(blocks), dim3(BLOCK_THREADS),
                                            0, stream>>>(
      species, structure_offsets, all_species, out, n_structures, n_atoms,
      n_species);
}
